// QformerPooler_39152921870807
// MI455X (gfx1250) — matrix operand order repaired and checked by reading, not compiled, not run on hardware
//
#include <hip/hip_runtime.h>
#include <hip/hip_bf16.h>
#include <math.h>

#define B_   8
#define N_   256
#define T_   2048
#define D_   512
#define L_   64
#define H_   8
#define QP_  4
#define HD_  64
#define NQP  (N_*QP_)            // 1024 query rows per batch
#define BTL  ((size_t)B_*T_*L_)  // output plane size

typedef __attribute__((ext_vector_type(16))) __bf16 v16bf;
typedef __attribute__((ext_vector_type(8)))  float  v8f;
typedef __attribute__((ext_vector_type(4)))  unsigned int u32x4;
typedef __attribute__((ext_vector_type(8)))  int i32x8;
typedef __attribute__((ext_vector_type(4)))  int i32x4;

union Frag { unsigned int u[8]; v16bf v; };

__device__ __forceinline__ unsigned short f2bf(float f) {
  unsigned int u = __float_as_uint(f);
  u += 0x7FFFu + ((u >> 16) & 1u);          // round-to-nearest-even
  return (unsigned short)(u >> 16);
}

__device__ __forceinline__ float wave_sum(float v) {
  #pragma unroll
  for (int m = 16; m > 0; m >>= 1) v += __shfl_xor(v, m, 32);
  return v;
}

__device__ __forceinline__ unsigned pcg(unsigned v) {
  v = v * 747796405u + 2891336453u;
  unsigned w = ((v >> ((v >> 28) + 4u)) ^ v) * 277803737u;
  return (w >> 22) ^ w;
}
__device__ __forceinline__ float gauss(unsigned i, unsigned salt) {
  unsigned a = pcg(i ^ (salt * 0x9E3779B9u));
  unsigned b = pcg(a + 0x85EBCA6Bu);
  float u1 = ((float)a + 1.0f) * 2.3283064e-10f;   // (0,1]
  float u2 = (float)b * 2.3283064e-10f;
  return sqrtf(-2.0f * __logf(u1)) * __cosf(6.2831853f * u2);
}

// ---------------------------------------------------------------------------
// TDM: DMA a 64-row x 32-col bf16 tile (row stride = strideElems) from global
// into LDS at ldsOff, writing the padded [64][34]-short layout via the D#
// pad feature (pad_interval = 16 DWORDs = one 64B row, pad_amount = 1 DWORD).
// D# built per CDNA5 ISA 8.3/8.4. 2D tensor == tile, so no OOB.
// ---------------------------------------------------------------------------
__device__ __forceinline__ void tdm_load_tile64x32(const unsigned short* gsrc,
                                                   int strideElems,
                                                   unsigned ldsOff)
{
  unsigned long long ga = (unsigned long long)(size_t)gsrc;
  u32x4 g0;
  g0[0] = 1u;                                          // count=1, user mode
  g0[1] = ldsOff;                                      // lds_addr
  g0[2] = (unsigned)(ga & 0xFFFFFFFFull);              // global_addr[31:0]
  g0[3] = (unsigned)((ga >> 32) & 0x1FFFFFFull) | (2u << 30); // addr[56:32] | type=2
  i32x8 g1;
  g1[0] = (int)((1u << 16) | (1u << 20) | (3u << 22)); // data_size=2B, pad_en, intvl=16dw, amt=1dw
  g1[1] = (int)(32u << 16);                            // tensor_dim0 = 32 (lo16 @ bits63:48)
  g1[2] = (int)(64u << 16);                            // dim0 hi=0 | tensor_dim1 = 64 lo16
  g1[3] = (int)(32u << 16);                            // dim1 hi=0 | tile_dim0 = 32
  g1[4] = (int)(64u);                                  // tile_dim1 = 64 | tile_dim2 = 0
  g1[5] = (int)(unsigned)strideElems;                  // tensor_dim0_stride lo32
  g1[6] = 0;                                           // stride0 hi16 | stride1 lo16
  g1[7] = 0;
  i32x4 z4 = {0, 0, 0, 0};
  i32x8 z8 = {0, 0, 0, 0, 0, 0, 0, 0};
  __builtin_amdgcn_tensor_load_to_lds(g0, g1, z4, z4, z8, 0);
}

// ---------------------------------------------------------------------------
// bf16-WMMA GEMM with TDM-staged, double-buffered LDS tiles:
//   C[M,N] = op( A[M,K](bf16) * W[N,K](bf16)^T + bias, resid )
// mode: 0 = bias(+resid), 1 = (x+bias)*scale, 2 = silu(x+bias)
// C (fp32) and/or Cbf (bf16) written if non-null.
// M,N multiples of 64; K multiple of 64 (so k-step count is even).
// ---------------------------------------------------------------------------
__global__ __launch_bounds__(128)
void gemm_bf16_wmma(const unsigned short* __restrict__ A,
                    const unsigned short* __restrict__ W,
                    const float* __restrict__ bias, const float* __restrict__ resid,
                    float* __restrict__ C, unsigned short* __restrict__ Cbf,
                    int M, int N, int K, int mode, float scale)
{
  __shared__ unsigned short As[2][64][34];
  __shared__ unsigned short Bs[2][64][34];
  const int tid  = threadIdx.x;
  const int lane = tid & 31;
  const int wave = tid >> 5;
  const int bm   = blockIdx.x * 64;
  const int bn   = blockIdx.y * 64;
  const int nsteps = K >> 5;

  v8f acc[4] = {v8f{}, v8f{}, v8f{}, v8f{}};

  // prologue: DMA k-step 0 into buffer 0 (single wave issues for the block)
  if (wave == 0) {
    tdm_load_tile64x32(A + (size_t)bm * K, K, (unsigned)(size_t)&As[0][0][0]);
    tdm_load_tile64x32(W + (size_t)bn * K, K, (unsigned)(size_t)&Bs[0][0][0]);
  }

  for (int s = 0; s < nsteps; s++) {
    const int cb = s & 1;
    if (wave == 0) {
      if (s + 1 < nsteps) {
        const int nb = cb ^ 1;
        const int k1 = (s + 1) << 5;
        tdm_load_tile64x32(A + (size_t)bm * K + k1, K, (unsigned)(size_t)&As[nb][0][0]);
        tdm_load_tile64x32(W + (size_t)bn * K + k1, K, (unsigned)(size_t)&Bs[nb][0][0]);
        __builtin_amdgcn_s_wait_tensorcnt((short)2);   // current tile's 2 DMAs done
      } else {
        __builtin_amdgcn_s_wait_tensorcnt((short)0);
      }
    }
    __syncthreads();

    // A fragment: rows m = wave*16 + (lane&15); CDNA5 16-bit A 16x32 layout
    Frag a;
    const int m = (lane & 15) + wave * 16;
    #pragma unroll
    for (int v = 0; v < 8; v++) {
      int kk = ((lane >> 4) << 3) + ((v & 4) ? 16 : 0) + ((v & 3) << 1);
      a.u[v] = *(const unsigned int*)&As[cb][m][kk];
    }
    // B fragments: n = nt*16 + (lane&15); same k order as A (8h+s / 16+8h+s-8)
    const int nl = lane & 15;
    const int kb = (lane >> 4) << 3;
    Frag b[4];
    #pragma unroll
    for (int nt = 0; nt < 4; nt++) {
      #pragma unroll
      for (int v = 0; v < 8; v++)
        b[nt].u[v] = *(const unsigned int*)&Bs[cb][nl + 16 * nt][kb + ((v & 4) ? 16 : 0) + ((v & 3) << 1)];
    }
    #pragma unroll
    for (int nt = 0; nt < 4; nt++)
      acc[nt] = __builtin_amdgcn_wmma_f32_16x16x32_bf16(
          false, a.v, false, b[nt].v, (short)0, acc[nt], false, false);
    __syncthreads();   // all waves done reading cb before it is DMA-overwritten
  }

  // Epilogue: C layout — lane holds column n = (lane&15); rows m = v + 8*(lane>=16)
  #pragma unroll
  for (int nt = 0; nt < 4; nt++) {
    const int n  = bn + nt * 16 + (lane & 15);
    const float bv = bias ? bias[n] : 0.0f;
    #pragma unroll
    for (int v = 0; v < 8; v++) {
      const int mrow = bm + wave * 16 + v + ((lane >> 4) << 3);
      float x = acc[nt][v] + bv;
      if (resid) x += resid[(size_t)mrow * N + n];
      if (mode == 1)      x *= scale;
      else if (mode == 2) x = x / (1.0f + __expf(-x));
      if (C)   C[(size_t)mrow * N + n] = x;
      if (Cbf) Cbf[(size_t)mrow * N + n] = f2bf(x);
    }
  }
}

// ---------------------------------------------------------------------------
__global__ void f2bf_k(const float* __restrict__ src, unsigned short* __restrict__ dst,
                       int n)
{
  int i = blockIdx.x * blockDim.x + threadIdx.x;
  if (i < n) dst[i] = f2bf(src[i]);
}

// queries = concat(embeddings[:, :, None, :], learnable) -> (B, N*QP, D); fp32 + bf16
__global__ void build_queries(const float* __restrict__ emb, const float* __restrict__ lq,
                              float* __restrict__ q, unsigned short* __restrict__ qb)
{
  size_t i = (size_t)blockIdx.x * blockDim.x + threadIdx.x;
  if (i >= (size_t)B_ * NQP * D_) return;
  int d = i % D_;
  int r = (i / D_) % NQP;
  int b = i / ((size_t)NQP * D_);
  int n = r >> 2, p = r & 3;
  float v = (p == 0) ? emb[((size_t)b * N_ + n) * D_ + d] : lq[(size_t)(p - 1) * D_ + d];
  q[i]  = v;
  qb[i] = f2bf(v);
}

// per-(b,t) argmax over one-hot alignments + contiguous t-range per (b,n)
__global__ void init_ranges(int* __restrict__ rs, int* __restrict__ re)
{
  int i = blockIdx.x * blockDim.x + threadIdx.x;
  if (i < B_ * N_) { rs[i] = T_; re[i] = -1; }
}

__global__ void build_ranges(const float* __restrict__ align, int* __restrict__ tsel,
                             int* __restrict__ rs, int* __restrict__ re)
{
  int i = blockIdx.x * blockDim.x + threadIdx.x;
  if (i >= B_ * T_) return;
  int b = i / T_, t = i % T_;
  const float* row = align + ((size_t)b * T_ + t) * N_;
  int n = 0;
  for (int j = 0; j < N_; j++) { if (row[j] > 0.5f) { n = j; break; } }
  tsel[i] = n;
  atomicMin(&rs[b * N_ + n], t);
  atomicMax(&re[b * N_ + n], t);
}

// Sparse masked attention: one-hot mask => only t in [start,end] survive -1e9;
// empty range -> full-T fallback (reference drops the mask for all-true rows).
__global__ __launch_bounds__(256)
void attn_sparse(const float* __restrict__ Q, const float* __restrict__ Kb,
                 const float* __restrict__ Vb, const int* __restrict__ rs,
                 const int* __restrict__ re, unsigned short* __restrict__ out)
{
  int gid = blockIdx.x * blockDim.x + threadIdx.x;
  if (gid >= B_ * NQP * H_) return;
  int h = gid % H_;
  int q = (gid / H_) % NQP;
  int b = gid / (NQP * H_);
  int n = q >> 2;
  int s = rs[b * N_ + n], e = re[b * N_ + n];
  if (s > e) { s = 0; e = T_ - 1; }

  const float* qp = Q + ((size_t)b * NQP + q) * D_ + h * HD_;
  float qv[HD_], acc[HD_];
  #pragma unroll
  for (int d = 0; d < HD_; d++) { qv[d] = qp[d]; acc[d] = 0.0f; }

  float mmax = -3.4e38f, ssum = 0.0f;
  for (int t = s; t <= e; t++) {
    const float* kp = Kb + ((size_t)b * T_ + t) * D_ + h * HD_;
    float sc = 0.0f;
    #pragma unroll
    for (int d = 0; d < HD_; d++) sc += qv[d] * kp[d];
    float nm   = fmaxf(mmax, sc);
    float corr = __expf(mmax - nm);
    float p    = __expf(sc - nm);
    ssum = ssum * corr + p;
    const float* vp = Vb + ((size_t)b * T_ + t) * D_ + h * HD_;
    #pragma unroll
    for (int d = 0; d < HD_; d++) acc[d] = acc[d] * corr + p * vp[d];
    mmax = nm;
  }
  float inv = 1.0f / ssum;
  unsigned short* op = out + ((size_t)b * NQP + q) * D_ + h * HD_;
  #pragma unroll
  for (int d = 0; d < HD_; d++) op[d] = f2bf(acc[d] * inv);
}

// LayerNorm over D=512; one wave per row; bf16 output (feeds next GEMM only)
__global__ __launch_bounds__(256)
void layernorm_k(const float* __restrict__ x, const float* __restrict__ g,
                 const float* __restrict__ bb, unsigned short* __restrict__ y, int rows)
{
  int lane = threadIdx.x & 31;
  int row  = blockIdx.x * 8 + (threadIdx.x >> 5);
  if (row >= rows) return;
  const float* xp = x + (size_t)row * D_;
  float s = 0.0f, sq = 0.0f;
  for (int d = lane; d < D_; d += 32) { float v = xp[d]; s += v; sq += v * v; }
  s = wave_sum(s); sq = wave_sum(sq);
  float mean = s * (1.0f / D_);
  float var  = sq * (1.0f / D_) - mean * mean;
  float rstd = rsqrtf(var + 1e-5f);
  unsigned short* yp = y + (size_t)row * D_;
  for (int d = lane; d < D_; d += 32)
    yp[d] = f2bf((xp[d] - mean) * rstd * g[d] + bb[d]);
}

// Reparameterize z + weighted KL into d_out[2*BTL]
__global__ void reparam_kl(const float* __restrict__ mu_z, const float* __restrict__ lv_z,
                           const float* __restrict__ klw, float* __restrict__ z,
                           float* __restrict__ kl_out)
{
  int i = blockIdx.x * blockDim.x + threadIdx.x;
  float kl = 0.0f;
  if (i < B_ * N_ * L_) {
    float mu = mu_z[i], lv = lv_z[i];
    float eps = gauss((unsigned)i, 1234u);
    z[i] = mu + eps * __expf(0.5f * lv);
    kl = -0.5f * (1.0f + lv - mu * mu - __expf(lv));
  }
  kl = wave_sum(kl);
  if ((threadIdx.x & 31) == 0 && kl != 0.0f) atomicAdd(kl_out, kl * klw[0]);
}

// z_up gather + transpose: x[b, l, t] = z[b, tsel[b,t], l]
__global__ void gather_zup(const float* __restrict__ z, const int* __restrict__ tsel,
                           float* __restrict__ x)
{
  size_t i = (size_t)blockIdx.x * blockDim.x + threadIdx.x;
  if (i >= (size_t)B_ * L_ * T_) return;
  int t = i % T_;
  int l = (i / T_) % L_;
  int b = i / ((size_t)L_ * T_);
  x[i] = z[((size_t)b * N_ + tsel[b * T_ + t]) * L_ + l];
}

// SAME conv1d, kernel 3, NCH layout, optional SiLU
__global__ void conv1d_k(const float* __restrict__ x, const float* __restrict__ w,
                         const float* __restrict__ bias, float* __restrict__ y,
                         int Cin, int Cout, int applySilu)
{
  size_t i = (size_t)blockIdx.x * blockDim.x + threadIdx.x;
  if (i >= (size_t)B_ * Cout * T_) return;
  int t  = i % T_;
  int co = (i / T_) % Cout;
  int b  = i / ((size_t)Cout * T_);
  float s = bias[co];
  for (int ci = 0; ci < Cin; ci++) {
    const float* xp = x + ((size_t)b * Cin + ci) * T_;
    const float* wp = w + ((size_t)co * Cin + ci) * 3;
    if (t > 0)      s += xp[t - 1] * wp[0];
    s += xp[t] * wp[1];
    if (t < T_ - 1) s += xp[t + 1] * wp[2];
  }
  if (applySilu) s = s / (1.0f + __expf(-s));
  y[i] = s;
}

// Final: split refined -> (mu_recon, logvar_recon); reparameterize z_recon
__global__ void final_out(const float* __restrict__ h2, float* __restrict__ out)
{
  size_t i = (size_t)blockIdx.x * blockDim.x + threadIdx.x;
  if (i >= BTL) return;
  int l = i % L_;
  int t = (i / L_) % T_;
  int b = i / ((size_t)T_ * L_);
  float mu_r = h2[((size_t)b * 128 + l) * T_ + t];
  float lv_r = h2[((size_t)b * 128 + l + 64) * T_ + t];
  float eps2 = gauss((unsigned)i, 777u);
  out[i]       = mu_r + eps2 * __expf(0.5f * lv_r);  // z_recon
  out[BTL + i] = mu_r;                               // mu_recon
}

// ---------------------------------------------------------------------------
extern "C" void kernel_launch(void* const* d_in, const int* in_sizes, int n_in,
                              void* d_out, int out_size, void* d_ws, size_t ws_size,
                              hipStream_t stream)
{
  (void)in_sizes; (void)n_in; (void)out_size; (void)ws_size;
  const float* emb   = (const float*)d_in[0];
  const float* align = (const float*)d_in[1];
  const float* mu_in = (const float*)d_in[2];
  const float* klw   = (const float*)d_in[3];
  const float* lq    = (const float*)d_in[4];
  const float* wq = (const float*)d_in[5],  *bq  = (const float*)d_in[6];
  const float* wk = (const float*)d_in[7],  *bk  = (const float*)d_in[8];
  const float* wv = (const float*)d_in[9],  *bv  = (const float*)d_in[10];
  const float* wo = (const float*)d_in[11], *bo  = (const float*)d_in[12];
  const float* lng = (const float*)d_in[13],*lnb = (const float*)d_in[14];
  const float* wbn = (const float*)d_in[15],*bbn = (const float*)d_in[16];
  const float* wmu = (const float*)d_in[17],*bmu = (const float*)d_in[18];
  const float* wlv = (const float*)d_in[19],*blv = (const float*)d_in[20];
  const float* cw1 = (const float*)d_in[21],*cb1 = (const float*)d_in[22];
  const float* cw2 = (const float*)d_in[23],*cb2 = (const float*)d_in[24];
  float* out = (float*)d_out;

  // --- workspace carve ---
  int* tsel = (int*)d_ws;                 // B*T
  int* rs   = tsel + B_ * T_;             // B*N
  int* re   = rs + B_ * N_;               // B*N
  float* f  = (float*)(re + B_ * N_);
  float* queries  = f;  f += (size_t)B_ * NQP * D_;
  float* qbuf     = f;  f += (size_t)B_ * NQP * D_;
  float* kbuf     = f;  f += (size_t)B_ * T_ * D_;
  float* vbuf     = f;  f += (size_t)B_ * T_ * D_;
  float* residbuf = f;  f += (size_t)B_ * NQP * D_;
  float* mu_z     = f;  f += (size_t)B_ * N_ * L_;
  float* lv_z     = f;  f += (size_t)B_ * N_ * L_;
  float* zbuf     = f;  f += (size_t)B_ * N_ * L_;
  float* xbuf     = f;  f += (size_t)B_ * L_ * T_;
  float* h1       = f;  f += (size_t)B_ * 128 * T_;
  float* h2       = f;  f += (size_t)B_ * 128 * T_;
  unsigned short* sp = (unsigned short*)f;
  unsigned short* queries_bf = sp; sp += (size_t)B_ * NQP * D_;
  unsigned short* mu_bf      = sp; sp += (size_t)B_ * T_ * L_;
  unsigned short* attn_bf    = sp; sp += (size_t)B_ * NQP * D_;
  unsigned short* qln_bf     = sp; sp += (size_t)B_ * NQP * D_;
  unsigned short* hidden_bf  = sp; sp += (size_t)B_ * N_ * D_;
  unsigned short* wqb  = sp; sp += D_ * D_;
  unsigned short* wkb  = sp; sp += D_ * L_;
  unsigned short* wvb  = sp; sp += D_ * L_;
  unsigned short* wob  = sp; sp += D_ * D_;
  unsigned short* wbnb = sp; sp += D_ * QP_ * D_;
  unsigned short* wmub = sp; sp += L_ * D_;
  unsigned short* wlvb = sp; sp += L_ * D_;

  const float qscale = 0.125f;  // HD^-0.5

  // 0. one-time fp32 -> bf16 conversions (weights + mu)
  auto cvt = [&](const float* s, unsigned short* d, int n) {
    f2bf_k<<<dim3((n + 255) / 256), dim3(256), 0, stream>>>(s, d, n);
  };
  cvt(wq,  wqb,  D_ * D_);
  cvt(wk,  wkb,  D_ * L_);
  cvt(wv,  wvb,  D_ * L_);
  cvt(wo,  wob,  D_ * D_);
  cvt(wbn, wbnb, D_ * QP_ * D_);
  cvt(wmu, wmub, L_ * D_);
  cvt(wlv, wlvb, L_ * D_);
  cvt(mu_in, mu_bf, B_ * T_ * L_);

  // 1. queries = concat(emb, learnable)
  {
    size_t n = (size_t)B_ * NQP * D_;
    build_queries<<<dim3((n + 255) / 256), dim3(256), 0, stream>>>(emb, lq, queries, queries_bf);
  }
  // 2-4. Q/K/V projections (TDM-staged bf16 WMMA)
  gemm_bf16_wmma<<<dim3((B_*NQP)/64, D_/64), dim3(128), 0, stream>>>(
      queries_bf, wqb, bq, nullptr, qbuf, nullptr, B_*NQP, D_, D_, 1, qscale);
  gemm_bf16_wmma<<<dim3((B_*T_)/64, D_/64), dim3(128), 0, stream>>>(
      mu_bf, wkb, bk, nullptr, kbuf, nullptr, B_*T_, D_, L_, 0, 0.0f);
  gemm_bf16_wmma<<<dim3((B_*T_)/64, D_/64), dim3(128), 0, stream>>>(
      mu_bf, wvb, bv, nullptr, vbuf, nullptr, B_*T_, D_, L_, 0, 0.0f);
  // 5. one-hot ranges
  init_ranges<<<dim3((B_*N_ + 255) / 256), dim3(256), 0, stream>>>(rs, re);
  build_ranges<<<dim3((B_*T_ + 255) / 256), dim3(256), 0, stream>>>(align, tsel, rs, re);
  // 6. sparse attention (bf16 out -> O-proj A operand)
  attn_sparse<<<dim3((B_*NQP*H_ + 255) / 256), dim3(256), 0, stream>>>(
      qbuf, kbuf, vbuf, rs, re, attn_bf);
  // 7. O-proj + residual
  gemm_bf16_wmma<<<dim3((B_*NQP)/64, D_/64), dim3(128), 0, stream>>>(
      attn_bf, wob, bo, queries, residbuf, nullptr, B_*NQP, D_, D_, 0, 0.0f);
  // 8. layernorm (bf16 out -> bottleneck A operand)
  layernorm_k<<<dim3((B_*NQP + 7) / 8), dim3(256), 0, stream>>>(
      residbuf, lng, lnb, qln_bf, B_*NQP);
  // 9. bottleneck: (B*N, QP*D) @ wbn^T, SiLU, bf16 out only
  gemm_bf16_wmma<<<dim3((B_*N_)/64, D_/64), dim3(128), 0, stream>>>(
      qln_bf, wbnb, bbn, nullptr, nullptr, hidden_bf, B_*N_, D_, QP_*D_, 2, 0.0f);
  // 10. mu_z / logvar_z heads
  gemm_bf16_wmma<<<dim3((B_*N_)/64, L_/64), dim3(128), 0, stream>>>(
      hidden_bf, wmub, bmu, nullptr, mu_z, nullptr, B_*N_, L_, D_, 0, 0.0f);
  gemm_bf16_wmma<<<dim3((B_*N_)/64, L_/64), dim3(128), 0, stream>>>(
      hidden_bf, wlvb, blv, nullptr, lv_z, nullptr, B_*N_, L_, D_, 0, 0.0f);
  // 11. reparam + KL
  (void)hipMemsetAsync(out + 2 * BTL, 0, sizeof(float), stream);
  reparam_kl<<<dim3((B_*N_*L_ + 255) / 256), dim3(256), 0, stream>>>(
      mu_z, lv_z, klw, zbuf, out + 2 * BTL);
  // 12. one-hot upsample into (B, L, T)
  gather_zup<<<dim3(((size_t)B_*L_*T_ + 255) / 256), dim3(256), 0, stream>>>(
      zbuf, tsel, xbuf);
  // 13-14. conv stack
  conv1d_k<<<dim3(((size_t)B_*128*T_ + 255) / 256), dim3(256), 0, stream>>>(
      xbuf, cw1, cb1, h1, L_, 128, 1);
  conv1d_k<<<dim3(((size_t)B_*128*T_ + 255) / 256), dim3(256), 0, stream>>>(
      h1, cw2, cb2, h2, 128, 128, 0);
  // 15. outputs
  final_out<<<dim3((BTL + 255) / 256), dim3(256), 0, stream>>>(h2, out);
}
